// DeepHedgingLoss_32890859553366
// MI455X (gfx1250) — matrix operand order repaired and checked by reading, not compiled, not run on hardware
//
#include <hip/hip_runtime.h>

typedef _Float16 f16;
typedef __attribute__((ext_vector_type(16))) _Float16 v16h;
typedef __attribute__((ext_vector_type(8)))  float    v8f;
typedef __attribute__((ext_vector_type(4)))  float    v4f;

#define N_STEPS 256
#define BATCH   131072
#define STRIKE  3.0f

#define WAVES_PER_BLOCK 4
#define THREADS        (WAVES_PER_BLOCK * 32)
#define PATHS_PER_BLOCK (WAVES_PER_BLOCK * 32)

__device__ __forceinline__ float fast_sig(float x) {
    // sigmoid(x) = 1 / (1 + e^-x); v_exp_f32 + v_rcp_f32
    float e = __expf(-x);
    return __builtin_amdgcn_rcpf(1.0f + e);
}

__global__ __launch_bounds__(THREADS)
void deep_hedge_milstein_wmma(const float* __restrict__ noise,
                              const float* __restrict__ ts,
                              const float* __restrict__ wref,
                              const float* __restrict__ W1,
                              const float* __restrict__ b1,
                              const float* __restrict__ W2,
                              const float* __restrict__ b2,
                              const float* __restrict__ W3,
                              const float* __restrict__ b3,
                              float* __restrict__ out)
{
    // Per-wave LDS: activation staging tile [path][0..31 val | 32..63 der] (f16),
    // plus layer-3 partial-sum transpose tile [path][col] -> (val, der).
    __shared__ __align__(16) f16    stage[WAVES_PER_BLOCK][32][64];
    __shared__ __align__(16) float2 red  [WAVES_PER_BLOCK][32][16];

    const int lane = threadIdx.x & 31;
    const int wv   = threadIdx.x >> 5;
    const int p    = blockIdx.x * PATHS_PER_BLOCK + wv * 32 + lane; // path owned by this lane
    const int half = lane >> 4;   // 0: lanes 0-15, 1: lanes 16-31
    const int col  = lane & 15;

    // ---- Persistent B fragments: W2^T, two 32x16 N-tiles, f16 (built once) ----
    // Dense-B layout: lanes 0-15 hold column `col`, K in {0..7, 16..23};
    // lanes 16-31 hold column `col`, K in {8..15, 24..31}.
    v16h bfr[2];
    {
        const int kb = half * 8;
        #pragma unroll
        for (int nt = 0; nt < 2; ++nt) {
            const int j = col + 16 * nt;            // output feature (W2 row)
            #pragma unroll
            for (int i = 0; i < 16; ++i)
                bfr[nt][i] = (f16)W2[j * 32 + kb + i + (i & 8)];
        }
    }
    const float b2v[2] = { b2[col], b2[col + 16] };
    const float w3v[2] = { W3[col], W3[col + 16] };
    const float b3v    = b3[0];

    float s = 1.0f, pnl = 0.0f;

    for (int step = 0; step < N_STEPS; ++step) {
        // Issue coalesced noise load early; consumed only at end of step.
        const float nz = noise[step * BATCH + p];
        const float t  = ts[step];
        const float dt = ts[step + 1] - t;
        const float dW = nz * sqrtf(dt);

        // ---------- Layer 1 (scalar per lane; K=2) ----------
        // z1_j = W1[j][0]*t + W1[j][1]*s + b1[j];  dz1_j/ds = W1[j][1]
        union { f16 h[32]; v4f f[4]; } av, ad;
        #pragma unroll
        for (int j = 0; j < 32; ++j) {
            const float w0 = W1[2 * j], w1 = W1[2 * j + 1];
            const float z  = fmaf(w0, t, fmaf(w1, s, b1[j]));
            const float sg = fast_sig(z);
            av.h[j] = (f16)(z * sg);                               // silu(z)
            ad.h[j] = (f16)((sg + z * sg * (1.0f - sg)) * w1);     // silu'(z)*dz/ds
        }
        {   // contiguous 128-bit LDS stores: row [path][feat]
            v4f* dv = (v4f*)&stage[wv][lane][0];
            v4f* dd = (v4f*)&stage[wv][lane][32];
            #pragma unroll
            for (int q = 0; q < 4; ++q) { dv[q] = av.f[q]; dd[q] = ad.f[q]; }
        }
        __syncthreads();

        // ---------- Layer 2: A-fragment loads (ISA 16-bit A 16x32 layout) ----------
        // lane<16: row M=col, K in {0..7, 16..23}; lane>=16: row M=col, K in {8..15, 24..31}
        v16h Av[2], Ad[2];
        {
            const int kb = half * 8;
            #pragma unroll
            for (int mt = 0; mt < 2; ++mt) {
                const f16* rp = &stage[wv][mt * 16 + col][0];
                union { v4f f4[2]; v16h h; } uv, ud;
                uv.f4[0] = *(const v4f*)(rp + kb);
                uv.f4[1] = *(const v4f*)(rp + kb + 16);
                ud.f4[0] = *(const v4f*)(rp + 32 + kb);
                ud.f4[1] = *(const v4f*)(rp + 32 + kb + 16);
                Av[mt] = uv.h;
                Ad[mt] = ud.h;
            }
        }

        // ---------- 8x V_WMMA_F32_16X16X32_F16: value and ds-derivative matvecs ----------
        v8f Dv[2][2] = {{{0}}}, Dd[2][2] = {{{0}}};
        #pragma unroll
        for (int mt = 0; mt < 2; ++mt) {
            #pragma unroll
            for (int nt = 0; nt < 2; ++nt) {
                Dv[mt][nt] = __builtin_amdgcn_wmma_f32_16x16x32_f16(
                    false, Av[mt], false, bfr[nt], (short)0, Dv[mt][nt], false, false);
                Dd[mt][nt] = __builtin_amdgcn_wmma_f32_16x16x32_f16(
                    false, Ad[mt], false, bfr[nt], (short)0, Dd[mt][nt], false, false);
            }
        }

        // ---------- Layer-2 SiLU + layer-3 pre-multiplied partials on D-fragments ----------
        // D layout: VGPR v -> row M = v + 8*half (per M-tile), column N = col.
        #pragma unroll
        for (int mt = 0; mt < 2; ++mt) {
            #pragma unroll
            for (int v = 0; v < 8; ++v) {
                float pv = 0.0f, pd = 0.0f;
                #pragma unroll
                for (int nt = 0; nt < 2; ++nt) {
                    const float z  = Dv[mt][nt][v] + b2v[nt];
                    const float sg = fast_sig(z);
                    const float a2 = z * sg;                                   // silu
                    const float d2 = (sg + z * sg * (1.0f - sg)) * Dd[mt][nt][v];
                    pv = fmaf(a2, w3v[nt], pv);
                    pd = fmaf(d2, w3v[nt], pd);
                }
                const int path = mt * 16 + v + 8 * half;
                red[wv][path][col] = make_float2(pv, pd);   // ds_store_b64
            }
        }
        __syncthreads();

        // ---------- Layer-3 reduction: lane gathers its own path's row ----------
        float zv = 0.0f, zd = 0.0f;
        #pragma unroll
        for (int c = 0; c < 16; ++c) {
            const float2 pr = red[wv][lane][c];   // vectorizes to ds_read_b128
            zv += pr.x; zd += pr.y;
        }

        const float sg3 = fast_sig(zv + b3v);     // h = sigmoid(z3)
        const float h   = sg3;
        const float dh  = sg3 * (1.0f - sg3) * zd;// dh/ds

        // ---------- Milstein step (SIGMA = MU = 1) ----------
        // f = [s, s*h]; g = [s, s*h]; dg = JVP(diffusion)(g) = [s, (h + s*dh)*s]
        const float mil = 0.5f * (dW * dW - dt);
        const float dg1 = (h + s * dh) * s;
        const float s_new   = s   + s * dt     + s * dW     + s   * mil;
        const float pnl_new = pnl + s * h * dt + s * h * dW + dg1 * mil;
        s = s_new; pnl = pnl_new;
    }

    const float w = wref[0];
    const float z = fmaxf(0.0f, s - STRIKE);
    const float d = z - pnl - w;
    out[p] = d * d;
}

extern "C" void kernel_launch(void* const* d_in, const int* in_sizes, int n_in,
                              void* d_out, int out_size, void* d_ws, size_t ws_size,
                              hipStream_t stream) {
    const float* noise = (const float*)d_in[0];
    const float* ts    = (const float*)d_in[1];
    const float* w     = (const float*)d_in[2];
    const float* W1    = (const float*)d_in[3];
    const float* b1    = (const float*)d_in[4];
    const float* W2    = (const float*)d_in[5];
    const float* b2    = (const float*)d_in[6];
    const float* W3    = (const float*)d_in[7];
    const float* b3    = (const float*)d_in[8];
    float* out = (float*)d_out;

    dim3 grid(BATCH / PATHS_PER_BLOCK);
    dim3 block(THREADS);
    deep_hedge_milstein_wmma<<<grid, block, 0, stream>>>(
        noise, ts, w, W1, b1, W2, b2, W3, b3, out);
}
